// VectorQuantizer_17265768529944
// MI455X (gfx1250) — matrix operand order repaired and checked by reading, not compiled, not run on hardware
//
#include <hip/hip_runtime.h>
#include <hip/hip_bf16.h>

typedef __attribute__((ext_vector_type(16))) __bf16 v16bf;
typedef __attribute__((ext_vector_type(8)))  float  v8f;
typedef __attribute__((ext_vector_type(4)))  float  v4f;

#define WMMA_BF16(a, b, c)                                                    \
  __builtin_amdgcn_wmma_f32_16x16x32_bf16(false, (a), false, (b), (short)0,   \
                                          (c), false, false)

// Problem constants (fixed by the reference: B=32, C=D=64, H=W=64, K=512)
constexpr int D_DIM = 64;        // embedding dim
constexpr int K_CB  = 512;       // codebook entries
constexpr int HW    = 4096;      // H*W
constexpr int CHW   = 64 * 4096; // C*H*W
constexpr int BLOCK_THREADS  = 128;           // 4 waves
constexpr int ROWS_PER_WAVE  = 32;            // two 16-row stripes share B
constexpr int ROWS_PER_BLOCK = 4 * ROWS_PER_WAVE; // 128
constexpr int LDS_STRIDE = D_DIM + 1;         // 65 -> conflict-free transpose

// ---------------------------------------------------------------------------
// Prep: split emb into bf16 hi/lo pair and compute ||e_k||^2 (f32, exact).
// ---------------------------------------------------------------------------
__global__ void vq_prep(const float* __restrict__ emb,
                        __bf16* __restrict__ ehi,
                        __bf16* __restrict__ elo,
                        float* __restrict__ e2) {
  int k = blockIdx.x * blockDim.x + threadIdx.x;
  if (k >= K_CB) return;
  const float* row = emb + k * D_DIM;
  float s = 0.0f;
#pragma unroll 8
  for (int c = 0; c < D_DIM; ++c) {
    float v = row[c];
    s = __builtin_fmaf(v, v, s);
    __bf16 h = (__bf16)v;                 // RNE round to bf16
    __bf16 l = (__bf16)(v - (float)h);    // residual
    ehi[k * D_DIM + c] = h;
    elo[k * D_DIM + c] = l;
  }
  e2[k] = s;
}

// ---------------------------------------------------------------------------
// Main: per block, 128 x-rows vs all 512 codes. 4 waves/block, each wave
// owns TWO 16-row stripes that share B fragments -> 12 WMMAs per 8 b128
// loads, two independent accumulator chains for XDL pipelining.
// bf16 hi/lo 3-term split gives ~fp32-accurate dot products (f32 accum).
// ---------------------------------------------------------------------------
__global__ __launch_bounds__(BLOCK_THREADS)
void vq_main(const float* __restrict__ x,
             const float* __restrict__ emb,
             const __bf16* __restrict__ ehi,
             const __bf16* __restrict__ elo,
             const float* __restrict__ e2,
             float* __restrict__ out) {
  __shared__ float xs[ROWS_PER_BLOCK * LDS_STRIDE];

  const int n0  = blockIdx.x * ROWS_PER_BLOCK;   // first flat row
  const int b   = n0 >> 12;                      // n0 / HW
  const int hw0 = n0 & (HW - 1);
  const int tid = threadIdx.x;

  // Coalesced load of x tile (NCHW: contiguous along n for fixed c),
  // transposed into LDS as [row][c] with padded stride.
  const float* xb = x + (size_t)b * CHW + hw0;
#pragma unroll
  for (int it = 0; it < (ROWS_PER_BLOCK * D_DIM) / BLOCK_THREADS; ++it) {
    int f = it * BLOCK_THREADS + tid;
    int c = f >> 7;    // f / 128
    int r = f & 127;   // row within tile
    xs[r * LDS_STRIDE + c] = xb[c * HW + r];
  }
  __syncthreads();

  const int wave = tid >> 5;   // 0..3
  const int lane = tid & 31;
  const int m    = lane & 15;  // A row / B col / C col
  const int half = lane >> 4;  // lane-half select

  // Build A fragments (hi/lo bf16 split) for this wave's 2 stripes x 16 rows.
  // A 16x32 layout: lane (m,half); elem e -> K = half*16 + e
  v16bf a_hi[2][2], a_lo[2][2];   // [stripe][kstep]
#pragma unroll
  for (int r = 0; r < 2; ++r) {
    const float* xrow = xs + (wave * ROWS_PER_WAVE + r * 16 + m) * LDS_STRIDE;
#pragma unroll
    for (int s = 0; s < 2; ++s) {
#pragma unroll
      for (int e = 0; e < 16; ++e) {
        int k = s * 32 + half * 16 + e;
        float v  = xrow[k];
        __bf16 h = (__bf16)v;
        a_hi[r][s][e] = h;
        a_lo[r][s][e] = (__bf16)(v - (float)h);
      }
    }
  }

  float bestVal[2][8];
  int   bestIdx[2][8];
#pragma unroll
  for (int r = 0; r < 2; ++r)
#pragma unroll
    for (int v = 0; v < 8; ++v) { bestVal[r][v] = 3.4e38f; bestIdx[r][v] = 0; }

  // 32 code tiles of 16 codes; 12 WMMAs per tile (2 K-steps x 3 terms x 2 stripes).
#pragma unroll 2
  for (int t = 0; t < K_CB / 16; ++t) {
    const int col = t * 16 + m;
    // B layout (32x16 K-major): lane col = m, elem e -> K = half*16 + e
    const __bf16* bh0 = ehi + col * D_DIM + half * 16;
    const __bf16* bl0 = elo + col * D_DIM + half * 16;
    v16bf bh[2], bl[2];
    bh[0] = *(const v16bf*)(bh0);
    bh[1] = *(const v16bf*)(bh0 + 32);
    bl[0] = *(const v16bf*)(bl0);
    bl[1] = *(const v16bf*)(bl0 + 32);
    float e2v = e2[col];

    v8f acc0 = {}, acc1 = {};
#pragma unroll
    for (int s = 0; s < 2; ++s) {
      // interleave the two stripes' accumulator chains
      acc0 = WMMA_BF16(a_hi[0][s], bh[s], acc0);  // hi*hi
      acc1 = WMMA_BF16(a_hi[1][s], bh[s], acc1);
      acc0 = WMMA_BF16(a_hi[0][s], bl[s], acc0);  // hi*lo
      acc1 = WMMA_BF16(a_hi[1][s], bl[s], acc1);
      acc0 = WMMA_BF16(a_lo[0][s], bh[s], acc0);  // lo*hi
      acc1 = WMMA_BF16(a_lo[1][s], bh[s], acc1);
    }

#pragma unroll
    for (int v = 0; v < 8; ++v) {
      float s0 = __builtin_fmaf(acc0[v], -2.0f, e2v); // ||e||^2 - 2 x.e
      float s1 = __builtin_fmaf(acc1[v], -2.0f, e2v);
      if (s0 < bestVal[0][v]) { bestVal[0][v] = s0; bestIdx[0][v] = col; }
      if (s1 < bestVal[1][v]) { bestVal[1][v] = s1; bestIdx[1][v] = col; }
    }
  }

  // Cross-lane argmin within each 16-lane half (rows 8*half .. 8*half+7).
#pragma unroll
  for (int mask = 1; mask <= 8; mask <<= 1) {
#pragma unroll
    for (int r = 0; r < 2; ++r) {
#pragma unroll
      for (int v = 0; v < 8; ++v) {
        float ov = __shfl_xor(bestVal[r][v], mask, 32);
        int   oi = __shfl_xor(bestIdx[r][v], mask, 32);
        if (ov < bestVal[r][v] ||
            (ov == bestVal[r][v] && oi < bestIdx[r][v])) {
          bestVal[r][v] = ov;
          bestIdx[r][v] = oi;
        }
      }
    }
  }

  // Gather + store: row = n0 + wave*32 + r*16 + 8*half + v;
  // lane m writes cols 4m..4m+3 (256B contiguous per row across 16 lanes).
#pragma unroll
  for (int r = 0; r < 2; ++r) {
#pragma unroll
    for (int v = 0; v < 8; ++v) {
      int row = n0 + wave * ROWS_PER_WAVE + r * 16 + half * 8 + v;
      int idx = bestIdx[r][v];
      v4f q = *(const v4f*)(emb + idx * D_DIM + m * 4);
      *(v4f*)(out + (size_t)row * D_DIM + m * 4) = q;
    }
  }
}

// ---------------------------------------------------------------------------
extern "C" void kernel_launch(void* const* d_in, const int* in_sizes, int n_in,
                              void* d_out, int out_size, void* d_ws,
                              size_t ws_size, hipStream_t stream) {
  const float* x   = (const float*)d_in[0];  // [32,64,64,64] f32
  const float* emb = (const float*)d_in[1];  // [512,64] f32
  float* out = (float*)d_out;                // [131072,64] f32

  // Workspace: emb_hi bf16 (64KB) | emb_lo bf16 (64KB) | e2 f32 (2KB)
  __bf16* ehi = (__bf16*)d_ws;
  __bf16* elo = ehi + K_CB * D_DIM;
  float*  e2  = (float*)((char*)d_ws + 2 * K_CB * D_DIM * sizeof(__bf16));

  vq_prep<<<dim3(2), dim3(256), 0, stream>>>(emb, ehi, elo, e2);

  const int N = 32 * HW;                       // 131072 rows
  vq_main<<<dim3(N / ROWS_PER_BLOCK), dim3(BLOCK_THREADS), 0, stream>>>(
      x, emb, ehi, elo, e2, out);
}
